// Seq2Seq_2284922601566
// MI455X (gfx1250) — matrix operand order repaired and checked by reading, not compiled, not run on hardware
//
#include <hip/hip_runtime.h>

// ---------------------------------------------------------------------------
// Seq2Seq 3-layer LSTM (B=8192, T=512, D=8, H=64) for gfx1250 / MI455X.
// 4 waves per block: each wave owns one {i,f,g,o} 16-column gate group of a
// 16-row batch tile and runs the whole enc+dec recurrence with
// v_wmma_f32_16x16x32_f16. LDS-resident state, L2-resident repacked weights,
// software-pipelined B-fragment loads, round-robin accumulators (no WMMA RAW).
// ---------------------------------------------------------------------------

typedef __attribute__((ext_vector_type(16))) _Float16 v16h;
typedef __attribute__((ext_vector_type(8)))  _Float16 v8h;
typedef __attribute__((ext_vector_type(8)))  float    v8f;

constexpr int TLEN = 512;
constexpr int BSZ  = 8192;

// LDS input-buffer strides (in halves); rows stay 16-byte aligned.
constexpr int S0 = 168;   // layer-0 buffer: K capacity 160 (dec0 needs 136->pad 160)
constexpr int S1 = 136;   // layers 1/2:     K capacity 128

// Weight-fragment offsets in d_ws (units: halves). One fragment = 32 lanes x 16 halves.
constexpr size_t OFF_E0 = 0;              // 16 ntiles * 3 ktiles = 48 frags
constexpr size_t OFF_E1 = 48ull  * 512;   // 64 frags
constexpr size_t OFF_E2 = 112ull * 512;   // 64 frags
constexpr size_t OFF_D0 = 176ull * 512;   // 80 frags
constexpr size_t OFF_D1 = 256ull * 512;   // 64 frags
constexpr size_t OFF_D2 = 320ull * 512;   // 64 frags
constexpr size_t OFF_FC = 384ull * 512;   // 2  frags

struct SeqParams {
  const float* enc_x; const float* dec_x;
  const float* be0; const float* be1; const float* be2;
  const float* bd0; const float* bd1; const float* bd2;
  const float* fc_b;
  const _Float16* wf;     // repacked f16 weight fragments in d_ws
  float* out;
};

// ---------------------------------------------------------------------------
// Prep: repack W = [Wih | Whh] (K-concatenated, zero padded) into B-fragment
// order: dst[(frag*32 + lane)*16 + j] = W[nt*16+(lane&15)][kt*32+8*(lane>>4)+(j<8?j:j+8)].
// ---------------------------------------------------------------------------
__global__ void prep_kernel(const float* __restrict__ Wih,
                            const float* __restrict__ Whh,
                            int fan, int ktiles,
                            _Float16* __restrict__ dst, int nrows)
{
  const int frag = blockIdx.x;          // frag = nt*ktiles + kt
  const int kt   = frag % ktiles;
  const int nt   = frag / ktiles;
  const int lane = threadIdx.x;
  const int n    = nt * 16 + (lane & 15);
#pragma unroll
  for (int j = 0; j < 16; ++j) {
    const int k = kt * 32 + ((lane >> 4) << 3) + (j < 8 ? j : j + 8);
    float v = 0.0f;
    if (n < nrows) {
      if (k < fan)                               v = Wih[(size_t)n * fan + k];
      else if (Whh != nullptr && k < fan + 64)   v = Whh[(size_t)n * 64 + (k - fan)];
    }
    dst[((size_t)frag * 32 + lane) * 16 + j] = (_Float16)v;
  }
}

// ---------------------------------------------------------------------------
// Device helpers
// ---------------------------------------------------------------------------
__device__ __forceinline__ float sigmf(float x) {
  return 1.0f / (1.0f + __expf(-x));
}
__device__ __forceinline__ float tanhfast(float x) {
  float e = __expf(2.0f * x);
  return (e - 1.0f) / (e + 1.0f);
}

// 16-bit A-matrix 16x32 fragment from row-major f16 LDS:
// lane L: M = L&15; halves 0..7 -> K = k0 + (L>=16?8:0) + j,
//         halves 8..15 -> K = k0 + 16 + (L>=16?8:0) + (j-8).
__device__ __forceinline__ v16h load_a_frag(const _Float16* base, int stride,
                                            int lane, int k0)
{
  const int m  = lane & 15;
  const int kb = (lane >> 4) << 3;
  const _Float16* p = base + m * stride + k0 + kb;
  v8h lo = *(const v8h*)(p);
  v8h hi = *(const v8h*)(p + 16);
  return __builtin_shufflevector(lo, hi, 0,1,2,3,4,5,6,7,8,9,10,11,12,13,14,15);
}

// ---------------------------------------------------------------------------
// One LSTM cell, executed cooperatively by 4 waves: wave `hn` computes gate
// columns n in [hn*16, hn*16+16) for all four gates {i,f,g,o}.
//  - 4 independent accumulators, round-robin in the K loop (no WMMA RAW chain)
//  - B fragments software-pipelined one K-step ahead
// ---------------------------------------------------------------------------
template <int KT>
__device__ __forceinline__ void lstm_cell(
    const _Float16* __restrict__ wf,   // fragment base for this weight set
    const float*    __restrict__ bias, // fused b_ih+b_hh, [256] in i,f,g,o order
    const _Float16* inb, int sIn,      // LDS input buffer [16][sIn], K = KT*32
    float* cp,                         // this wave's c slice (8 floats per lane)
    _Float16* hA, int sA, int cA,      // recurrence slot (always written)
    _Float16* hB, int sB, int cB,      // next-layer input slot (optional)
    _Float16* hC, int sC, int cC,      // decoder y_prev slot   (optional)
    int hn, int lane)
{
  v16h a[KT];
#pragma unroll
  for (int kt = 0; kt < KT; ++kt) a[kt] = load_a_frag(inb, sIn, lane, kt * 32);
  __syncthreads();   // every wave has captured A before any wave writes h below

  const v16h* bf = (const v16h*)wf;
  v8f acc[4] = {};

  v16h bcur[4], bnxt[4];
#pragma unroll
  for (int g = 0; g < 4; ++g) bcur[g] = bf[((g * 4 + hn) * KT) * 32 + lane];

#pragma unroll
  for (int kt = 0; kt < KT; ++kt) {
    if (kt + 1 < KT) {
#pragma unroll
      for (int g = 0; g < 4; ++g)
        bnxt[g] = bf[((g * 4 + hn) * KT + kt + 1) * 32 + lane];
    }
#pragma unroll
    for (int g = 0; g < 4; ++g) {
      acc[g] = __builtin_amdgcn_wmma_f32_16x16x32_f16(
                   false, a[kt], false, bcur[g], (short)0, acc[g], false, false);
    }
    if (kt + 1 < KT) {
#pragma unroll
      for (int g = 0; g < 4; ++g) bcur[g] = bnxt[g];
    }
  }

  const int nlo = lane & 15;
  const int mhi = (lane >> 4) << 3;
  const int n   = hn * 16 + nlo;
  const float bi  = bias[n];
  const float bfv = bias[64 + n];
  const float bg  = bias[128 + n];
  const float bo  = bias[192 + n];

#pragma unroll
  for (int r = 0; r < 8; ++r) {
    const float iv = sigmf(acc[0][r] + bi);
    const float fv = sigmf(acc[1][r] + bfv);
    const float gv = tanhfast(acc[2][r] + bg);
    const float ov = sigmf(acc[3][r] + bo);
    const float cn = fv * cp[r] + iv * gv;
    cp[r] = cn;
    const _Float16 h16 = (_Float16)(ov * tanhfast(cn));
    const int m = mhi + r;                   // C/D layout: M = r + 8*(lane>=16)
    hA[m * sA + cA + n] = h16;
    if (hB) hB[m * sB + cB + n] = h16;
    if (hC) hC[m * sC + cC + n] = h16;
  }
  __syncthreads();   // h visible to all waves before the next layer reads it
}

// ---------------------------------------------------------------------------
// Main kernel: grid = B/16 blocks, 128 threads (4 waves) each.
// ---------------------------------------------------------------------------
__global__ __launch_bounds__(128) void seq2seq_kernel(SeqParams p)
{
  __shared__ alignas(32) _Float16 ib0[16 * S0];   // [x | y_prev | h0] concat buf
  __shared__ alignas(32) _Float16 ib1[16 * S1];   // [h0_out | h1] concat buf
  __shared__ alignas(32) _Float16 ib2[16 * S1];   // [h1_out | h2] concat buf
  __shared__ alignas(32) float    cbuf[3 * 4 * 32 * 8];

  const int    tid  = threadIdx.x;
  const int    lane = tid & 31;
  const int    hn   = tid >> 5;          // wave id == gate column group
  const size_t b0   = (size_t)blockIdx.x * 16;

  // c-state slice owned exclusively by this (wave, lane)
  float* c0 = cbuf + 0    + ((size_t)(hn * 32 + lane)) * 8;
  float* c1 = cbuf + 1024 + ((size_t)(hn * 32 + lane)) * 8;
  float* c2 = cbuf + 2048 + ((size_t)(hn * 32 + lane)) * 8;

  // zero state + padding
  for (int i = tid; i < 16 * S0; i += 128) ib0[i] = (_Float16)0.0f;
  for (int i = tid; i < 16 * S1; i += 128) { ib1[i] = (_Float16)0.0f; ib2[i] = (_Float16)0.0f; }
  for (int i = tid; i < 3 * 4 * 32 * 8; i += 128) cbuf[i] = 0.0f;
  __syncthreads();

  const int srow = tid >> 3;            // 16 rows x 8 cols: one value per thread
  const int scol = tid & 7;

  // ---------------- encoder ----------------
  for (int t = 0; t < TLEN; ++t) {
    const float xv = p.enc_x[(((b0 + srow) * TLEN + t) << 3) + scol];
    ib0[srow * S0 + scol] = (_Float16)xv;
    __syncthreads();

    // l0: K = [x(8) | h0(64)] pad 96
    lstm_cell<3>(p.wf + OFF_E0, p.be0, ib0, S0, c0,
                 ib0, S0, 8,  ib1, S1, 0,  (_Float16*)nullptr, 0, 0, hn, lane);
    // l1: K = [h0(64) | h1(64)]
    lstm_cell<4>(p.wf + OFF_E1, p.be1, ib1, S1, c1,
                 ib1, S1, 64, ib2, S1, 0,  (_Float16*)nullptr, 0, 0, hn, lane);
    // l2: K = [h1(64) | h2(64)]
    lstm_cell<4>(p.wf + OFF_E2, p.be2, ib2, S1, c2,
                 ib2, S1, 64, (_Float16*)nullptr, 0, 0, (_Float16*)nullptr, 0, 0, hn, lane);
  }

  // ---------------- enc -> dec transition ----------------
  // decoder l0 input layout: [dec_x(8) | y_prev(64) | h0(64)]; move h0 from
  // cols 8..71 (enc layout) to 72..135, zero y_prev (y0 = 0).
  // Each thread copies then zeroes its own elements -> no barrier needed between.
  for (int i = tid; i < 16 * 64; i += 128) {
    const int m = i >> 6, n = i & 63;
    ib0[m * S0 + 72 + n] = ib0[m * S0 + 8 + n];
  }
  for (int i = tid; i < 16 * 64; i += 128) {
    const int m = i >> 6, n = i & 63;
    ib0[m * S0 + 8 + n] = (_Float16)0.0f;
  }
  __syncthreads();

  // ---------------- decoder ----------------
  const v16h* fcf = (const v16h*)(p.wf + OFF_FC);

  for (int t = 0; t < TLEN; ++t) {
    const float xv = p.dec_x[(((b0 + srow) * TLEN + t) << 3) + scol];
    ib0[srow * S0 + scol] = (_Float16)xv;
    __syncthreads();

    // l0: K = [x(8) | y_prev(64) | h0(64)] pad 160
    lstm_cell<5>(p.wf + OFF_D0, p.bd0, ib0, S0, c0,
                 ib0, S0, 72, ib1, S1, 0,  (_Float16*)nullptr, 0, 0, hn, lane);
    lstm_cell<4>(p.wf + OFF_D1, p.bd1, ib1, S1, c1,
                 ib1, S1, 64, ib2, S1, 0,  (_Float16*)nullptr, 0, 0, hn, lane);
    // l2: recurrence slot + y_prev slot for next timestep
    lstm_cell<4>(p.wf + OFF_D2, p.bd2, ib2, S1, c2,
                 ib2, S1, 64, (_Float16*)nullptr, 0, 0, ib0, S0, 8, hn, lane);

    // fc: pred(16x8) = h2(16x64) @ fc_W^T  (one padded N-tile, 2 K-tiles).
    // Wave 0 only (uniform branch; EXEC all-ones inside the wave).
    if (hn == 0) {
      v16h a0 = load_a_frag(ib2 + 64, S1, lane, 0);
      v16h a1 = load_a_frag(ib2 + 64, S1, lane, 32);
      v16h w0 = fcf[lane];
      v16h w1 = fcf[32 + lane];
      v8f acc = {};
      acc = __builtin_amdgcn_wmma_f32_16x16x32_f16(false, a0, false, w0,
                                                   (short)0, acc, false, false);
      acc = __builtin_amdgcn_wmma_f32_16x16x32_f16(false, a1, false, w1,
                                                   (short)0, acc, false, false);
      const int nlo = lane & 15;
      const int mhi = (lane >> 4) << 3;
      if (nlo < 8) {
        const float bb = p.fc_b[nlo];
#pragma unroll
        for (int r = 0; r < 8; ++r) {
          p.out[(((b0 + mhi + r) * TLEN + t) << 3) + nlo] = acc[r] + bb;
        }
      }
    }
    // No barrier needed here: other waves proceed to next-t staging (ib0 x-part),
    // which does not touch ib2; the post-staging barrier re-converges everyone.
  }
}

// ---------------------------------------------------------------------------
// Launch
// ---------------------------------------------------------------------------
extern "C" void kernel_launch(void* const* d_in, const int* in_sizes, int n_in,
                              void* d_out, int out_size, void* d_ws, size_t ws_size,
                              hipStream_t stream)
{
  (void)in_sizes; (void)n_in; (void)out_size; (void)ws_size;

  const float* enc_x = (const float*)d_in[0];
  const float* dec_x = (const float*)d_in[1];
  const float* eW0 = (const float*)d_in[2];
  const float* eU0 = (const float*)d_in[3];
  const float* eb0 = (const float*)d_in[4];
  const float* eW1 = (const float*)d_in[5];
  const float* eU1 = (const float*)d_in[6];
  const float* eb1 = (const float*)d_in[7];
  const float* eW2 = (const float*)d_in[8];
  const float* eU2 = (const float*)d_in[9];
  const float* eb2 = (const float*)d_in[10];
  const float* dW0 = (const float*)d_in[11];
  const float* dU0 = (const float*)d_in[12];
  const float* db0 = (const float*)d_in[13];
  const float* dW1 = (const float*)d_in[14];
  const float* dU1 = (const float*)d_in[15];
  const float* db1 = (const float*)d_in[16];
  const float* dW2 = (const float*)d_in[17];
  const float* dU2 = (const float*)d_in[18];
  const float* db2 = (const float*)d_in[19];
  const float* fcW = (const float*)d_in[20];
  const float* fcb = (const float*)d_in[21];

  _Float16* wf = (_Float16*)d_ws;   // ~386 KB of repacked weight fragments

  prep_kernel<<<16 * 3, 32, 0, stream>>>(eW0, eU0,  8, 3, wf + OFF_E0, 256);
  prep_kernel<<<16 * 4, 32, 0, stream>>>(eW1, eU1, 64, 4, wf + OFF_E1, 256);
  prep_kernel<<<16 * 4, 32, 0, stream>>>(eW2, eU2, 64, 4, wf + OFF_E2, 256);
  prep_kernel<<<16 * 5, 32, 0, stream>>>(dW0, dU0, 72, 5, wf + OFF_D0, 256);
  prep_kernel<<<16 * 4, 32, 0, stream>>>(dW1, dU1, 64, 4, wf + OFF_D1, 256);
  prep_kernel<<<16 * 4, 32, 0, stream>>>(dW2, dU2, 64, 4, wf + OFF_D2, 256);
  prep_kernel<<<2,      32, 0, stream>>>(fcW, nullptr, 64, 2, wf + OFF_FC, 8);

  SeqParams p;
  p.enc_x = enc_x; p.dec_x = dec_x;
  p.be0 = eb0; p.be1 = eb1; p.be2 = eb2;
  p.bd0 = db0; p.bd1 = db1; p.bd2 = db2;
  p.fc_b = fcb; p.wf = wf; p.out = (float*)d_out;

  seq2seq_kernel<<<BSZ / 16, 128, 0, stream>>>(p);
}
